// ObservationFusionModule_49512382988613
// MI455X (gfx1250) — compile-verified, matrix operand order checked by reading, not run on hardware
//
#include <hip/hip_runtime.h>
#include <hip/hip_bf16.h>
#include <math.h>

typedef __attribute__((ext_vector_type(16))) _Float16 v16h;
typedef __attribute__((ext_vector_type(8)))  _Float16 v8h;
typedef __attribute__((ext_vector_type(8)))  float    v8f;

#define S_STACK 4
#define H_DIM   256
#define WID     128
#define TM      16          // rows of L per workgroup
#define TR      64          // TM * S_STACK
#define NTHREADS 128
#define NWAVES   4

// ---------------- LDS layout (byte offsets into one arena) ----------------
#define LDMOD 264           // f16 row stride (256 + 8 pad -> 16B-aligned, bank skew)
#define LDCAT 776           // f16: [h(256) | fused(256) | q(256)] + 8 pad
#define LDX   136           // f16: MLP activations 128 + 8 pad
#define LDF   260           // f32: fused, 256 + 4 pad

#define OFF_MOD   0
#define OFF_CAT   (OFF_MOD + TR * LDMOD * 2)     // 33792
#define OFF_R     (OFF_CAT + TM * LDCAT * 2)     // 58624 : reused region
#define OFF_X1    OFF_R                          // MLP act 1 (early phases)
#define OFF_X2    (OFF_R + TR * LDX * 2)         // MLP act 2 (early phases)
#define OFF_FUSED OFF_R                          // fused f32 (late phases)
#define OFF_SCORE (OFF_R + 2 * TR * LDX * 2)     // 93440 : [16][8] f32
#define OFF_WT    (OFF_SCORE + TM * 8 * 4)       // [16][8] f32 softmax weights
#define OFF_STAT  (OFF_WT + TM * 8 * 4)          // [16][4] f32 mu/rstd/any
#define OFF_LOG   (OFF_STAT + TM * 4 * 4)        // [64] f32 log1p(staleness)
#define SMEM_BYTES (OFF_LOG + TR * 4)            // ~94976 B

// ---------------- workspace layout (f16 element offsets) ----------------
#define WS_W1 0
#define WS_W2 (WS_W1 + WID * WID)                // 16384
#define WS_WQ (WS_W2 + H_DIM * WID)              // 49152
#define WS_WK (WS_WQ + H_DIM * H_DIM)            // 114688
#define WS_WV (WS_WK + H_DIM * H_DIM)            // 180224
#define WS_WG (WS_WV + H_DIM * H_DIM)            // 245760 (size 256*512)

// Build a 16-bit WMMA fragment per the CDNA5 ISA lane/VGPR interleave:
// lane r=lane&15 owns row (A) / col (B); khalf=(lane>>4)*8 selects K-halves.
// Two contiguous 8-halfword (16B) chunks per 32-wide K step.
__device__ __forceinline__ v16h make_frag(const _Float16* __restrict__ rowp, int khalf) {
  const v8h lo = *(const v8h*)(rowp + khalf);
  const v8h hi = *(const v8h*)(rowp + 16 + khalf);
  v16h f;
#pragma unroll
  for (int i = 0; i < 8; ++i) { f[i] = lo[i]; f[i + 8] = hi[i]; }
  return f;
}

// K-loop with A fragments streamed from LDS (used when A has little reuse).
template<int K>
__device__ __forceinline__ v8f gemm_kloop(const _Float16* __restrict__ arow,
                                          const _Float16* __restrict__ wrow,
                                          int khalf, v8f c) {
#pragma unroll
  for (int kb = 0; kb < K; kb += 32) {
    const v16h a = make_frag(arow + kb, khalf);
    const v16h b = make_frag(wrow + kb, khalf);
    c = __builtin_amdgcn_wmma_f32_16x16x32_f16(false, a, false, b, (short)0, c,
                                               false, false);
  }
  return c;
}

// K-loop with A fragments pre-staged in registers (reused across all N tiles).
template<int NF>
__device__ __forceinline__ v8f gemm_kloop_pre(const v16h* __restrict__ aF,
                                              const _Float16* __restrict__ wrow,
                                              int khalf, v8f c) {
#pragma unroll
  for (int f = 0; f < NF; ++f) {
    const v16h b = make_frag(wrow + f * 32, khalf);
    c = __builtin_amdgcn_wmma_f32_16x16x32_f16(false, aF[f], false, b, (short)0, c,
                                               false, false);
  }
  return c;
}

__global__ void cvt_f32_to_f16(const float* __restrict__ src,
                               _Float16* __restrict__ dst, int n) {
  const int i = blockIdx.x * blockDim.x + threadIdx.x;
  if (i < n) dst[i] = (_Float16)src[i];
}

__global__ __launch_bounds__(NTHREADS, 1)
void obsfuse_kernel(const float* __restrict__ h, const float* __restrict__ obs,
                    const float* __restrict__ st, const unsigned char* __restrict__ valid,
                    const _Float16* __restrict__ wqh, const float* __restrict__ bq,
                    const _Float16* __restrict__ wkh, const float* __restrict__ bk,
                    const _Float16* __restrict__ wvh, const float* __restrict__ bv,
                    const float* __restrict__ nullk, const float* __restrict__ nullv,
                    const float* __restrict__ w0, const float* __restrict__ b0,
                    const _Float16* __restrict__ w1h, const float* __restrict__ b1,
                    const _Float16* __restrict__ w2h, const float* __restrict__ b2,
                    const _Float16* __restrict__ wgh, const float* __restrict__ bg,
                    const float* __restrict__ lnw, const float* __restrict__ lnb,
                    float* __restrict__ out) {
  __shared__ __align__(16) unsigned char smem[SMEM_BYTES];
  _Float16* MOD   = (_Float16*)(smem + OFF_MOD);
  _Float16* CAT   = (_Float16*)(smem + OFF_CAT);
  _Float16* X1    = (_Float16*)(smem + OFF_X1);
  _Float16* X2    = (_Float16*)(smem + OFF_X2);
  float*    FUSED = (float*)(smem + OFF_FUSED);
  float*    SCORE = (float*)(smem + OFF_SCORE);
  float*    WT    = (float*)(smem + OFF_WT);
  float*    STAT  = (float*)(smem + OFF_STAT);
  float*    LOGST = (float*)(smem + OFF_LOG);

  const int tid  = threadIdx.x;
  const int wave = tid >> 5;
  const int lane = tid & 31;
  const int r16  = lane & 15;
  const int hi8  = (lane >> 4) << 3;   // doubles as khalf
  const int l0   = blockIdx.x * TM;

  const float* hB   = h   + (size_t)l0 * H_DIM;
  const float* obsB = obs + (size_t)l0 * S_STACK * H_DIM;
  const float* stB  = st  + (size_t)l0 * S_STACK;
  const unsigned char* vB = valid + (size_t)l0 * S_STACK;

  // ---- ph0: prefetch the big HBM streams for this tile so their latency
  // overlaps the MLP WMMA phases (global_prefetch_b8).
  for (int off = tid * 256; off < TR * H_DIM * 4; off += NTHREADS * 256)
    __builtin_prefetch((const char*)obsB + off, 0, 3);
  for (int off = tid * 256; off < TM * H_DIM * 4; off += NTHREADS * 256)
    __builtin_prefetch((const char*)hB + off, 0, 3);

  // stage h as f16 into CAT[:, :256]; log1p(staleness); zero accumulators
  for (int e = tid; e < TM * H_DIM; e += NTHREADS)
    CAT[(e >> 8) * LDCAT + (e & 255)] = (_Float16)hB[e];
  if (tid < TR)      LOGST[tid] = log1pf(stB[tid]);
  if (tid < TM * 8)  SCORE[tid] = 0.f;
  if (tid < TM * 4)  STAT[tid]  = 0.f;
  __syncthreads();

  // ---- ph0b: staleness MLP layer0 (outer product, VALU) -> X1
  {
    const int w = tid;                      // NTHREADS == WID
    const float w0v = w0[w], b0v = b0[w];
    for (int rr = 0; rr < TR; ++rr)
      X1[rr * LDX + w] = (_Float16)fmaxf(LOGST[rr] * w0v + b0v, 0.f);
  }
  __syncthreads();

  // ---- ph1: MLP layer1 [64x128]x[128x128], relu -> X2 (WMMA)
  // wave owns M-block mt=wave; A fragments staged once, reused for all N tiles
  {
    v16h aF[WID / 32];
#pragma unroll
    for (int f = 0; f < WID / 32; ++f)
      aF[f] = make_frag(X1 + (wave * 16 + r16) * LDX + f * 32, hi8);
    for (int nt = 0; nt < WID / 16; ++nt) {
      v8f c = {};
      c = gemm_kloop_pre<WID / 32>(aF, w1h + (nt * 16 + r16) * WID, hi8, c);
      const int n = nt * 16 + r16, m0 = wave * 16 + hi8;
      const float b = b1[n];
#pragma unroll
      for (int j = 0; j < 8; ++j)
        X2[(m0 + j) * LDX + n] = (_Float16)fmaxf(c[j] + b, 0.f);
    }
  }
  __syncthreads();

  // ---- ph2: MLP layer2 [64x128]x[128x256] + obs -> MOD f16 (WMMA)
  {
    v16h aF[WID / 32];
#pragma unroll
    for (int f = 0; f < WID / 32; ++f)
      aF[f] = make_frag(X2 + (wave * 16 + r16) * LDX + f * 32, hi8);
    for (int nt = 0; nt < H_DIM / 16; ++nt) {
      v8f c = {};
      c = gemm_kloop_pre<WID / 32>(aF, w2h + (nt * 16 + r16) * WID, hi8, c);
      const int n = nt * 16 + r16, m0 = wave * 16 + hi8;
      const float b = b2[n];
#pragma unroll
      for (int j = 0; j < 8; ++j) {
        const int m = m0 + j;
        MOD[m * LDMOD + n] = (_Float16)(c[j] + b + obsB[m * H_DIM + n]);
      }
    }
  }
  __syncthreads();

  // ---- ph3: q GEMM [16x256]x[256x256] -> CAT[:, 512:768]; zero FUSED (X2 dead)
  for (int nt = wave; nt < H_DIM / 16; nt += NWAVES) {
    v8f c = {};
    c = gemm_kloop<H_DIM>(CAT + r16 * LDCAT, wqh + (nt * 16 + r16) * H_DIM, hi8, c);
    const int n = nt * 16 + r16;
    const float b = bq[n];
#pragma unroll
    for (int j = 0; j < 8; ++j)
      CAT[(hi8 + j) * LDCAT + 512 + n] = (_Float16)(c[j] + b);
  }
  for (int e = tid; e < TM * LDF; e += NTHREADS) FUSED[e] = 0.f;
  __syncthreads();

  // ---- ph4: K GEMM (WMMA); epilogue folds q·k into scores via ds_add_f32
  {
    v16h aF[H_DIM / 32];
#pragma unroll
    for (int f = 0; f < H_DIM / 32; ++f)
      aF[f] = make_frag(MOD + (wave * 16 + r16) * LDMOD + f * 32, hi8);
    for (int nt = 0; nt < H_DIM / 16; ++nt) {
      v8f c = {};
      c = gemm_kloop_pre<H_DIM / 32>(aF, wkh + (nt * 16 + r16) * H_DIM, hi8, c);
      const int n = nt * 16 + r16, m0 = wave * 16 + hi8;
      const float b = bk[n];
#pragma unroll
      for (int j = 0; j < 8; ++j) {
        const int m = m0 + j, l = m >> 2, s = m & 3;
        const float qv = (float)CAT[l * LDCAT + 512 + n];
        atomicAdd(&SCORE[l * 8 + s], qv * (c[j] + b));
      }
    }
  }
  { // null-key score partial dot (VALU)
    const int l = tid >> 3, c0 = (tid & 7) * 32;
    float acc = 0.f;
    for (int c = c0; c < c0 + 32; ++c)
      acc += (float)CAT[l * LDCAT + 512 + c] * nullk[c];
    atomicAdd(&SCORE[l * 8 + 4], acc);
  }
  __syncthreads();

  // ---- ph5: scale + validity mask + softmax over 5 keys
  if (tid < TM) {
    float sc[5], mx = -1e30f;
#pragma unroll
    for (int s = 0; s < 5; ++s) {
      float v = SCORE[tid * 8 + s] * 0.0625f;            // 1/sqrt(256)
      if (s < 4 && !vB[tid * 4 + s]) v -= 5.f;
      sc[s] = v; mx = fmaxf(mx, v);
    }
    float sum = 0.f;
#pragma unroll
    for (int s = 0; s < 5; ++s) { sc[s] = expf(sc[s] - mx); sum += sc[s]; }
    const float inv = 1.f / sum;
#pragma unroll
    for (int s = 0; s < 5; ++s) WT[tid * 8 + s] = sc[s] * inv;
  }
  __syncthreads();

  // ---- ph6: V GEMM (WMMA); epilogue folds weight*v into FUSED via ds_add_f32
  {
    v16h aF[H_DIM / 32];
#pragma unroll
    for (int f = 0; f < H_DIM / 32; ++f)
      aF[f] = make_frag(MOD + (wave * 16 + r16) * LDMOD + f * 32, hi8);
    for (int nt = 0; nt < H_DIM / 16; ++nt) {
      v8f c = {};
      c = gemm_kloop_pre<H_DIM / 32>(aF, wvh + (nt * 16 + r16) * H_DIM, hi8, c);
      const int n = nt * 16 + r16, m0 = wave * 16 + hi8;
      const float b = bv[n];
#pragma unroll
      for (int j = 0; j < 8; ++j) {
        const int m = m0 + j, l = m >> 2, s = m & 3;
        atomicAdd(&FUSED[l * LDF + n], WT[l * 8 + s] * (c[j] + b));
      }
    }
  }
  __syncthreads();

  // ---- ph7: add null-value, stash fused f16 for gate GEMM, LN partial sums
  {
    const int l = tid >> 3, c0 = (tid & 7) * 32;
    const float w4 = WT[l * 8 + 4];
    float sum = 0.f, sq = 0.f;
    for (int c = c0; c < c0 + 32; ++c) {
      const float f = FUSED[l * LDF + c] + w4 * nullv[c];
      FUSED[l * LDF + c] = f;
      CAT[l * LDCAT + 256 + c] = (_Float16)f;
      sum += f; sq += f * f;
    }
    atomicAdd(&STAT[l * 4 + 0], sum);
    atomicAdd(&STAT[l * 4 + 1], sq);
  }
  __syncthreads();

  // ---- ph8: finalize LN stats + any_real gate mask
  if (tid < TM) {
    const float mu  = STAT[tid * 4 + 0] * (1.f / H_DIM);
    const float var = STAT[tid * 4 + 1] * (1.f / H_DIM) - mu * mu;
    const unsigned char* vp = vB + tid * 4;
    STAT[tid * 4 + 0] = mu;
    STAT[tid * 4 + 1] = rsqrtf(var + 1e-5f);
    STAT[tid * 4 + 2] = (vp[0] | vp[1] | vp[2] | vp[3]) ? 1.f : 0.f;
  }
  __syncthreads();

  // ---- ph9: gate GEMM [16x512]x[512x256] (WMMA); sigmoid+LN+residual -> out
  for (int nt = wave; nt < H_DIM / 16; nt += NWAVES) {
    v8f c = {};
    c = gemm_kloop<2 * H_DIM>(CAT + r16 * LDCAT, wgh + (nt * 16 + r16) * (2 * H_DIM), hi8, c);
    const int n = nt * 16 + r16;
    const float b = bg[n], lw = lnw[n], lb = lnb[n];
#pragma unroll
    for (int j = 0; j < 8; ++j) {
      const int l = hi8 + j;
      const float g    = 1.f / (1.f + expf(-(c[j] + b)));
      const float mu   = STAT[l * 4 + 0];
      const float rstd = STAT[l * 4 + 1];
      const float any  = STAT[l * 4 + 2];
      const float upd  = (FUSED[l * LDF + n] - mu) * rstd * lw + lb;
      out[(size_t)(l0 + l) * H_DIM + n] = hB[l * H_DIM + n] + g * any * upd;
    }
  }
}

extern "C" void kernel_launch(void* const* d_in, const int* in_sizes, int n_in,
                              void* d_out, int out_size, void* d_ws, size_t ws_size,
                              hipStream_t stream) {
  const float* h     = (const float*)d_in[0];
  const float* obs   = (const float*)d_in[1];
  const float* st    = (const float*)d_in[2];
  const unsigned char* valid = (const unsigned char*)d_in[3];
  const float* Wq = (const float*)d_in[4];  const float* bq = (const float*)d_in[5];
  const float* Wk = (const float*)d_in[6];  const float* bk = (const float*)d_in[7];
  const float* Wv = (const float*)d_in[8];  const float* bv = (const float*)d_in[9];
  const float* nullk = (const float*)d_in[10];
  const float* nullv = (const float*)d_in[11];
  const float* w0 = (const float*)d_in[12]; const float* b0 = (const float*)d_in[13];
  const float* w1 = (const float*)d_in[14]; const float* b1 = (const float*)d_in[15];
  const float* w2 = (const float*)d_in[16]; const float* b2 = (const float*)d_in[17];
  const float* Wg = (const float*)d_in[18]; const float* bg = (const float*)d_in[19];
  const float* lnw = (const float*)d_in[20]; const float* lnb = (const float*)d_in[21];
  float* out = (float*)d_out;

  _Float16* ws  = (_Float16*)d_ws;
  _Float16* w1h = ws + WS_W1;
  _Float16* w2h = ws + WS_W2;
  _Float16* wqh = ws + WS_WQ;
  _Float16* wkh = ws + WS_WK;
  _Float16* wvh = ws + WS_WV;
  _Float16* wgh = ws + WS_WG;

  const int L = in_sizes[0] / H_DIM;

  // per-call weight f32->f16 conversion; weights stay L2-resident thereafter
  const int CB = 256;
  cvt_f32_to_f16<<<(WID * WID + CB - 1) / CB,     CB, 0, stream>>>(w1, w1h, WID * WID);
  cvt_f32_to_f16<<<(H_DIM * WID + CB - 1) / CB,   CB, 0, stream>>>(w2, w2h, H_DIM * WID);
  cvt_f32_to_f16<<<(H_DIM * H_DIM + CB - 1) / CB, CB, 0, stream>>>(Wq, wqh, H_DIM * H_DIM);
  cvt_f32_to_f16<<<(H_DIM * H_DIM + CB - 1) / CB, CB, 0, stream>>>(Wk, wkh, H_DIM * H_DIM);
  cvt_f32_to_f16<<<(H_DIM * H_DIM + CB - 1) / CB, CB, 0, stream>>>(Wv, wvh, H_DIM * H_DIM);
  cvt_f32_to_f16<<<(H_DIM * 2 * H_DIM + CB - 1) / CB, CB, 0, stream>>>(Wg, wgh, H_DIM * 2 * H_DIM);

  obsfuse_kernel<<<L / TM, NTHREADS, 0, stream>>>(
      h, obs, st, valid,
      wqh, bq, wkh, bk, wvh, bv, nullk, nullv,
      w0, b0, w1h, b1, w2h, b2, wgh, bg, lnw, lnb, out);
}
